// Attention_83966610637378
// MI455X (gfx1250) — matrix operand order repaired and checked by reading, not compiled, not run on hardware
//
#include <hip/hip_runtime.h>
#include <hip/hip_bf16.h>
#include <math.h>

// ---------------------------------------------------------------------------
// Shunted-attention block for MI455X (gfx1250, wave32, WMMA).
// All GEMM-shaped work (q proj, SR convs as im2col GEMM, kv proj, QK^T, PV,
// out proj) runs on v_wmma_f32_16x16x32_f16 (f16 in, f32 accumulate).
// LayerNorm / softmax / GELU statistics stay in f32 scalar math.
// ---------------------------------------------------------------------------

typedef __attribute__((ext_vector_type(16))) _Float16 v16h;
typedef __attribute__((ext_vector_type(8)))  _Float16 v8h;
typedef __attribute__((ext_vector_type(8)))  float    v8f;

#define Bsz   8
#define Cdim  256
#define Ntok  3144      // 3136 spatial + 8 msg
#define Npad  3152      // rounded to 16
#define HDim  32
#define ATT_SCALE 0.17677669529663687f   // 32^-0.5
#define LPMAX 800

static __device__ inline v8f wmma_f16(v16h a, v16h b, v8f c) {
  return __builtin_amdgcn_wmma_f32_16x16x32_f16(false, a, false, b, (short)0, c,
                                                false, false);
}

// A fragment (16x32, f16, row-major, row stride ld halves).
// Lane layout (ISA 7.12.2): row = lane&15, g = lane>>4; per lane the 16 halves
// are two contiguous runs: k in [g*8, g*8+8) and [16+g*8, 16+g*8+8).
static __device__ inline v16h a_frag_f16(const _Float16* p, int ld, int lane) {
  int row = lane & 15, g = lane >> 4;
  const _Float16* rp = p + (size_t)row * ld + g * 8;
  v8h lo = *(const v8h*)rp;
  v8h hi = *(const v8h*)(rp + 16);
  v16h a;
#pragma unroll
  for (int e = 0; e < 8; ++e) { a[e] = lo[e]; a[8 + e] = hi[e]; }
  return a;
}

// A fragment gathered from an f32 tile (e.g. softmax P in LDS), ld in floats.
static __device__ inline v16h a_frag_from_f32(const float* p, int ld, int lane) {
  int row = lane & 15, g = lane >> 4;
  const float* rp = p + (size_t)row * ld;
  v16h a;
#pragma unroll
  for (int e = 0; e < 16; ++e) {
    int k = (e < 8 ? e : e + 8) + g * 8;
    a[e] = (_Float16)rp[k];
  }
  return a;
}

// B fragment (32xK-tile x 16 cols) from K-major storage: element(col,k) at
// p[col*ldcol + k].  Lane: col = lane&15, g = lane>>4; k in [g*8, g*8+8) and
// [16+g*8, 16+g*8+8) -> two 16B loads per lane.
static __device__ inline v16h b_frag_f16(const _Float16* p, int ldcol, int lane) {
  int col = lane & 15, g = lane >> 4;
  const _Float16* rp = p + (size_t)col * ldcol + g * 8;
  v8h lo = *(const v8h*)rp;
  v8h hi = *(const v8h*)(rp + 16);
  v16h b;
#pragma unroll
  for (int e = 0; e < 8; ++e) { b[e] = lo[e]; b[8 + e] = hi[e]; }
  return b;
}

// B fragment gathered when K is the slow axis: element(col,k) at p[k*ldk+col].
static __device__ inline v16h b_frag_gather(const _Float16* p, int ldk, int lane) {
  int col = lane & 15, g = lane >> 4;
  v16h b;
#pragma unroll
  for (int e = 0; e < 16; ++e)
    b[e] = p[(size_t)((e < 8 ? e : e + 8) + g * 8) * ldk + col];
  return b;
}

// ---------------------------------------------------------------------------
// Small utility kernels
// ---------------------------------------------------------------------------
__global__ void zero_h(_Float16* p, int n) {
  for (int i = blockIdx.x * blockDim.x + threadIdx.x; i < n;
       i += gridDim.x * blockDim.x)
    p[i] = (_Float16)0.0f;
}

__global__ void f32_to_f16(const float* in, _Float16* out, int n) {
  for (int i = blockIdx.x * blockDim.x + threadIdx.x; i < n;
       i += gridDim.x * blockDim.x)
    out[i] = (_Float16)in[i];
}

// x_q = concat(x, msg) -> f16 [B, 3144, 256]
__global__ void build_xq(const float* x, const float* msg, _Float16* xq, int n) {
  for (int i = blockIdx.x * blockDim.x + threadIdx.x; i < n;
       i += gridDim.x * blockDim.x) {
    int c = i & 255;
    int rem = i >> 8;
    int t = rem % Ntok;
    int bz = rem / Ntok;
    float v = (t < 3136) ? x[((size_t)bz * 3136 + t) * 256 + c]
                         : msg[((size_t)bz * 8 + (t - 3136)) * 256 + c];
    xq[i] = (_Float16)v;
  }
}

// ---------------------------------------------------------------------------
// q = x_q @ q_w^T, stored f16 as [B, 8, Npad, 32]
// ---------------------------------------------------------------------------
__global__ __launch_bounds__(32) void qproj_kernel(const _Float16* xq,
                                                   const _Float16* wq,
                                                   _Float16* qh) {
  int lane = threadIdx.x;
  int row0 = blockIdx.x * 16, col0 = blockIdx.y * 16;
  int g = lane >> 4, cl = lane & 15;
  v8f acc = {};
  for (int kc = 0; kc < 256; kc += 32) {
    v16h a = a_frag_f16(xq + (size_t)row0 * 256 + kc, 256, lane);
    v16h b = b_frag_f16(wq + (size_t)col0 * 256 + kc, 256, lane);
    acc = wmma_f16(a, b, acc);
  }
  int col = col0 + cl;
  int h = col >> 5, d = col & 31;
#pragma unroll
  for (int r = 0; r < 8; ++r) {
    int n = row0 + r + 8 * g;
    int bz = n / Ntok, t = n - bz * Ntok;
    qh[(((size_t)bz * 8 + h) * Npad + t) * 32 + d] = (_Float16)acc[r];
  }
}

// ---------------------------------------------------------------------------
// SR branch conv as im2col GEMM, msg rows folded in (msg . sum_hw(w)).
// out_pre[n, col] = conv(+bias), f32.  RDIM = C*KH*KW.
// ---------------------------------------------------------------------------
template <int S, int KW, int KHW, int OW, int L0, int RDIM>
__global__ __launch_bounds__(32) void sr_conv_kernel(const float* x,
                                                     const float* msg,
                                                     const _Float16* wh,
                                                     const float* bias,
                                                     float* out) {
  const int Ltot = L0 + 8;
  int lane = threadIdx.x;
  int row0 = blockIdx.x * 16, col0 = blockIdx.y * 16;
  int g = lane >> 4, cl = lane & 15;
  int n = row0 + cl;
  int bz = n / Ltot, t = n - bz * Ltot;
  bool is_msg = (t >= L0);
  int oh = t / OW, ow = t - oh * OW;
  v8f acc = {};
  for (int kc = 0; kc < RDIM; kc += 32) {
    v16h a;
#pragma unroll
    for (int e = 0; e < 16; ++e) {
      int r = kc + (e < 8 ? e : e + 8) + g * 8;
      int c = r / KHW;
      int rem = r - c * KHW;
      int ii = rem / KW, jj = rem - (rem / KW) * KW;
      float v;
      if (is_msg)
        v = msg[((size_t)bz * 8 + (t - L0)) * 256 + c];
      else
        v = x[((size_t)bz * 3136 + (oh * S + ii) * 56 + (ow * S + jj)) * 256 + c];
      a[e] = (_Float16)v;
    }
    v16h b = b_frag_f16(wh + (size_t)col0 * RDIM + kc, RDIM, lane);
    acc = wmma_f16(a, b, acc);
  }
  int col = col0 + cl;
  float bs = bias[col];
#pragma unroll
  for (int r = 0; r < 8; ++r) {
    int nn = row0 + r + 8 * g;
    out[(size_t)nn * 256 + col] = acc[r] + bs;
  }
}

// ---------------------------------------------------------------------------
// LayerNorm (f32 stats) + exact GELU -> f16
// ---------------------------------------------------------------------------
__global__ __launch_bounds__(256) void ln_gelu_kernel(const float* xin,
                                                      const float* gw,
                                                      const float* bw,
                                                      _Float16* out) {
  __shared__ float red[8];
  int tok = blockIdx.x, c = threadIdx.x;
  float v = xin[(size_t)tok * 256 + c];
  float s = v;
#pragma unroll
  for (int m = 16; m >= 1; m >>= 1) s += __shfl_xor(s, m, 32);
  if ((c & 31) == 0) red[c >> 5] = s;
  __syncthreads();
  float mean = 0.f;
#pragma unroll
  for (int i = 0; i < 8; ++i) mean += red[i];
  mean *= (1.0f / 256.0f);
  float d = v - mean;
  float s2 = d * d;
#pragma unroll
  for (int m = 16; m >= 1; m >>= 1) s2 += __shfl_xor(s2, m, 32);
  __syncthreads();
  if ((c & 31) == 0) red[c >> 5] = s2;
  __syncthreads();
  float var = 0.f;
#pragma unroll
  for (int i = 0; i < 8; ++i) var += red[i];
  var *= (1.0f / 256.0f);
  float y = d * rsqrtf(var + 1e-5f) * gw[c] + bw[c];
  float ge = 0.5f * y * (1.0f + erff(y * 0.70710678118654752f));
  out[(size_t)tok * 256 + c] = (_Float16)ge;
}

// ---------------------------------------------------------------------------
// kv = x_sr @ kv_w^T  (f32 out buffer [rows, 256])
// ---------------------------------------------------------------------------
__global__ __launch_bounds__(32) void kv_gemm_kernel(const _Float16* xs,
                                                     const _Float16* w,
                                                     float* out) {
  int lane = threadIdx.x;
  int row0 = blockIdx.x * 16, col0 = blockIdx.y * 16;
  int g = lane >> 4, cl = lane & 15;
  v8f acc = {};
  for (int kc = 0; kc < 256; kc += 32) {
    v16h a = a_frag_f16(xs + (size_t)row0 * 256 + kc, 256, lane);
    v16h b = b_frag_f16(w + (size_t)col0 * 256 + kc, 256, lane);
    acc = wmma_f16(a, b, acc);
  }
  int col = col0 + cl;
#pragma unroll
  for (int r = 0; r < 8; ++r)
    out[(size_t)(row0 + r + 8 * g) * 256 + col] = acc[r];
}

// ---------------------------------------------------------------------------
// v-mix: depthwise 3x3 on spatial part of v, msg part doubled; writes padded
// f16 K/V buffers [B, 4, Lp, 32].
// ---------------------------------------------------------------------------
__global__ void vmix_kernel(const float* kv, const float* lcw, const float* lcb,
                            _Float16* kh, _Float16* vh, int Ltot, int L0,
                            int Hs, int Ws, int Lp) {
  int total = Bsz * 128 * Ltot;
  for (int i = blockIdx.x * blockDim.x + threadIdx.x; i < total;
       i += gridDim.x * blockDim.x) {
    int t = i % Ltot;
    int ch = (i / Ltot) % 128;
    int bz = i / (128 * Ltot);
    const float* base = kv + (size_t)bz * Ltot * 256;
    float vv = base[(size_t)t * 256 + 128 + ch];
    float kk = base[(size_t)t * 256 + ch];
    float mixed;
    if (t < L0) {
      int oh = t / Ws, ow = t - (t / Ws) * Ws;
      mixed = lcb[ch];
#pragma unroll
      for (int di = 0; di < 3; ++di)
#pragma unroll
        for (int dj = 0; dj < 3; ++dj) {
          int yy = oh + di - 1, xx = ow + dj - 1;
          if (yy >= 0 && yy < Hs && xx >= 0 && xx < Ws)
            mixed += lcw[ch * 9 + di * 3 + dj] *
                     base[(size_t)(yy * Ws + xx) * 256 + 128 + ch];
        }
    } else {
      mixed = vv;   // msg columns: mixed = v itself
    }
    int h = ch >> 5, d = ch & 31;
    size_t o = (((size_t)bz * 4 + h) * Lp + t) * 32 + d;
    vh[o] = (_Float16)(vv + mixed);
    kh[o] = (_Float16)kk;
  }
}

// ---------------------------------------------------------------------------
// Attention: one wave = 16 query rows x one head.  QK^T via WMMA (K=HD=32),
// exact f32 softmax in LDS, PV via WMMA reading P from LDS.
// ---------------------------------------------------------------------------
__global__ __launch_bounds__(32) void attn_kernel(const _Float16* qh,
                                                  const _Float16* kh,
                                                  const _Float16* vh, float* xo,
                                                  int hbase, int Lvalid,
                                                  int Lp) {
  __shared__ float sS[16 * LPMAX];
  __shared__ float sl[16];
  int lane = threadIdx.x;
  int row0 = blockIdx.x * 16;
  int h = blockIdx.y;
  int bz = blockIdx.z;
  int g = lane >> 4, cl = lane & 15;
  const _Float16* qb = qh + (((size_t)bz * 8 + hbase + h) * Npad + row0) * 32;
  const _Float16* kb = kh + (((size_t)bz * 4 + h) * Lp) * 32;
  const _Float16* vb = vh + (((size_t)bz * 4 + h) * Lp) * 32;
  v16h qf = a_frag_f16(qb, 32, lane);

  for (int ct = 0; ct < Lp; ct += 16) {
    v16h kf = b_frag_f16(kb + (size_t)ct * 32, 32, lane);
    v8f z = {};
    v8f s = wmma_f16(qf, kf, z);
    int col = ct + cl;
#pragma unroll
    for (int r = 0; r < 8; ++r)
      sS[(r + 8 * g) * LPMAX + col] =
          (col < Lvalid) ? s[r] * ATT_SCALE : -3.0e30f;
  }
  __syncthreads();

  int row = cl;
  int half = Lp >> 1;
  int tbeg = g * half, tend = tbeg + half;
  float m = -3.0e30f;
  for (int t = tbeg; t < tend; ++t) m = fmaxf(m, sS[row * LPMAX + t]);
  m = fmaxf(m, __shfl_xor(m, 16, 32));
  float sum = 0.f;
  for (int t = tbeg; t < tend; ++t) {
    float p = expf(sS[row * LPMAX + t] - m);
    sS[row * LPMAX + t] = p;
    sum += p;
  }
  sum += __shfl_xor(sum, 16, 32);
  if (lane < 16) sl[lane] = sum;
  __syncthreads();

  v8f o0 = {}, o1 = {};
  for (int kc = 0; kc < Lp; kc += 32) {
    v16h pf = a_frag_from_f32(sS + kc, LPMAX, lane);
    v16h v0 = b_frag_gather(vb + (size_t)kc * 32, 32, lane);
    v16h v1 = b_frag_gather(vb + (size_t)kc * 32 + 16, 32, lane);
    o0 = wmma_f16(pf, v0, o0);
    o1 = wmma_f16(pf, v1, o1);
  }
#pragma unroll
  for (int r = 0; r < 8; ++r) {
    int t = row0 + r + 8 * g;
    if (t < Ntok) {
      float inv = 1.0f / sl[r + 8 * g];
      float* op = xo + ((size_t)bz * Ntok + t) * 128 + h * 32;
      op[cl] = o0[r] * inv;
      op[16 + cl] = o1[r] * inv;
    }
  }
}

// ---------------------------------------------------------------------------
// out = concat(x1o, x2o) @ proj_w^T + proj_b + residual(x_q); write to d_out
// in the reference's tuple layout (spatial block first, msg block second).
// ---------------------------------------------------------------------------
__global__ __launch_bounds__(32) void proj_kernel(const float* x1o,
                                                  const float* x2o,
                                                  const _Float16* wp,
                                                  const float* pb,
                                                  const float* x,
                                                  const float* msg,
                                                  float* out) {
  int lane = threadIdx.x;
  int row0 = blockIdx.x * 16, col0 = blockIdx.y * 16;
  int g = lane >> 4, cl = lane & 15;
  int n = row0 + cl;
  v8f acc = {};
  for (int kc = 0; kc < 256; kc += 32) {
    v16h a;
#pragma unroll
    for (int e = 0; e < 16; ++e) {
      int k = kc + (e < 8 ? e : e + 8) + g * 8;
      float av = (k < 128) ? x1o[(size_t)n * 128 + k]
                           : x2o[(size_t)n * 128 + (k - 128)];
      a[e] = (_Float16)av;
    }
    v16h b = b_frag_f16(wp + (size_t)col0 * 256 + kc, 256, lane);
    acc = wmma_f16(a, b, acc);
  }
  int col = col0 + cl;
  float bias = pb[col];
#pragma unroll
  for (int r = 0; r < 8; ++r) {
    int nn = row0 + r + 8 * g;
    int bz = nn / Ntok, t = nn - bz * Ntok;
    float res, *dst;
    size_t oidx;
    if (t < 3136) {
      res = x[((size_t)bz * 3136 + t) * 256 + col];
      oidx = ((size_t)bz * 3136 + t) * 256 + col;
    } else {
      res = msg[((size_t)bz * 8 + (t - 3136)) * 256 + col];
      oidx = (size_t)8 * 3136 * 256 + ((size_t)bz * 8 + (t - 3136)) * 256 + col;
    }
    out[oidx] = acc[r] + bias + res;
  }
}

// ---------------------------------------------------------------------------
extern "C" void kernel_launch(void* const* d_in, const int* in_sizes, int n_in,
                              void* d_out, int out_size, void* d_ws,
                              size_t ws_size, hipStream_t stream) {
  const float* x       = (const float*)d_in[0];
  const float* msg     = (const float*)d_in[1];
  const float* q_w     = (const float*)d_in[2];
  const float* sr1_w   = (const float*)d_in[3];
  const float* sr1_b   = (const float*)d_in[4];
  const float* sr2_w   = (const float*)d_in[5];
  const float* sr2_b   = (const float*)d_in[6];
  const float* norm1_g = (const float*)d_in[7];
  const float* norm1_b = (const float*)d_in[8];
  const float* norm2_g = (const float*)d_in[9];
  const float* norm2_b = (const float*)d_in[10];
  const float* kv1_w   = (const float*)d_in[11];
  const float* kv2_w   = (const float*)d_in[12];
  const float* lc1_w   = (const float*)d_in[13];
  const float* lc1_b   = (const float*)d_in[14];
  const float* lc2_w   = (const float*)d_in[15];
  const float* lc2_b   = (const float*)d_in[16];
  const float* proj_w  = (const float*)d_in[17];
  const float* proj_b  = (const float*)d_in[18];
  float* outp = (float*)d_out;

  // workspace carve-up (256B aligned)
  unsigned char* ws = (unsigned char*)d_ws;
  size_t off = 0;
  auto take = [&](size_t bytes) -> void* {
    void* p = ws + off;
    off = (off + bytes + 255) & ~(size_t)255;
    return p;
  };
  const int L1 = 204, L2 = 792, L1p = 224, L2p = 800;
  _Float16* hw_q   = (_Float16*)take((size_t)65536 * 2);
  _Float16* hw_kv1 = (_Float16*)take((size_t)65536 * 2);
  _Float16* hw_kv2 = (_Float16*)take((size_t)65536 * 2);
  _Float16* hw_pr  = (_Float16*)take((size_t)65536 * 2);
  _Float16* hw_sr1 = (_Float16*)take((size_t)1048576 * 2);
  _Float16* hw_sr2 = (_Float16*)take((size_t)262144 * 2);
  _Float16* xq_h   = (_Float16*)take((size_t)Bsz * Ntok * 256 * 2);
  _Float16* q_h    = (_Float16*)take((size_t)Bsz * 8 * Npad * 32 * 2);
  float*    x1pre  = (float*)take((size_t)Bsz * L1 * 256 * 4);
  float*    x2pre  = (float*)take((size_t)Bsz * L2 * 256 * 4);
  _Float16* x1_h   = (_Float16*)take((size_t)Bsz * L1 * 256 * 2);
  _Float16* x2_h   = (_Float16*)take((size_t)Bsz * L2 * 256 * 2);
  float*    kv1    = (float*)take((size_t)Bsz * L1 * 256 * 4);
  float*    kv2    = (float*)take((size_t)Bsz * L2 * 256 * 4);
  _Float16* k1h    = (_Float16*)take((size_t)Bsz * 4 * L1p * 32 * 2);
  _Float16* v1h    = (_Float16*)take((size_t)Bsz * 4 * L1p * 32 * 2);
  _Float16* k2h    = (_Float16*)take((size_t)Bsz * 4 * L2p * 32 * 2);
  _Float16* v2h    = (_Float16*)take((size_t)Bsz * 4 * L2p * 32 * 2);
  float*    x1o    = (float*)take((size_t)Bsz * Ntok * 128 * 4);
  float*    x2o    = (float*)take((size_t)Bsz * Ntok * 128 * 4);

  // zero padded f16 buffers (pad rows must read as 0)
  zero_h<<<2048, 256, 0, stream>>>(q_h, Bsz * 8 * Npad * 32);
  zero_h<<<256, 256, 0, stream>>>(k1h, Bsz * 4 * L1p * 32);
  zero_h<<<256, 256, 0, stream>>>(v1h, Bsz * 4 * L1p * 32);
  zero_h<<<512, 256, 0, stream>>>(k2h, Bsz * 4 * L2p * 32);
  zero_h<<<512, 256, 0, stream>>>(v2h, Bsz * 4 * L2p * 32);

  // weight conversions f32 -> f16
  f32_to_f16<<<256, 256, 0, stream>>>(q_w, hw_q, 65536);
  f32_to_f16<<<256, 256, 0, stream>>>(kv1_w, hw_kv1, 65536);
  f32_to_f16<<<256, 256, 0, stream>>>(kv2_w, hw_kv2, 65536);
  f32_to_f16<<<256, 256, 0, stream>>>(proj_w, hw_pr, 65536);
  f32_to_f16<<<2048, 256, 0, stream>>>(sr1_w, hw_sr1, 1048576);
  f32_to_f16<<<1024, 256, 0, stream>>>(sr2_w, hw_sr2, 262144);

  build_xq<<<4096, 256, 0, stream>>>(x, msg, xq_h, Bsz * Ntok * 256);

  // q projection: rows 25152, cols 256
  qproj_kernel<<<dim3(1572, 16), 32, 0, stream>>>(xq_h, hw_q, q_h);

  // SR conv branches as im2col GEMMs (rows = B*(L0+8))
  sr_conv_kernel<4, 4, 16, 14, 196, 4096>
      <<<dim3(102, 16), 32, 0, stream>>>(x, msg, hw_sr1, sr1_b, x1pre);
  sr_conv_kernel<2, 2, 4, 28, 784, 1024>
      <<<dim3(396, 16), 32, 0, stream>>>(x, msg, hw_sr2, sr2_b, x2pre);

  ln_gelu_kernel<<<Bsz * L1, 256, 0, stream>>>(x1pre, norm1_g, norm1_b, x1_h);
  ln_gelu_kernel<<<Bsz * L2, 256, 0, stream>>>(x2pre, norm2_g, norm2_b, x2_h);

  kv_gemm_kernel<<<dim3(102, 16), 32, 0, stream>>>(x1_h, hw_kv1, kv1);
  kv_gemm_kernel<<<dim3(396, 16), 32, 0, stream>>>(x2_h, hw_kv2, kv2);

  vmix_kernel<<<816, 256, 0, stream>>>(kv1, lc1_w, lc1_b, k1h, v1h, L1, 196,
                                       14, 14, L1p);
  vmix_kernel<<<3168, 256, 0, stream>>>(kv2, lc2_w, lc2_b, k2h, v2h, L2, 784,
                                        28, 28, L2p);

  attn_kernel<<<dim3(Npad / 16, 4, Bsz), 32, 0, stream>>>(q_h, k1h, v1h, x1o,
                                                          0, L1, L1p);
  attn_kernel<<<dim3(Npad / 16, 4, Bsz), 32, 0, stream>>>(q_h, k2h, v2h, x2o,
                                                          4, L2, L2p);

  proj_kernel<<<dim3(1572, 16), 32, 0, stream>>>(x1o, x2o, hw_pr, proj_b, x,
                                                 msg, outp);
}
